// EncoderDecoder_57286273795012
// MI455X (gfx1250) — matrix operand order repaired and checked by reading, not compiled, not run on hardware
//
#include <hip/hip_runtime.h>
#include <hip/hip_bf16.h>
#include <cstdint>
#include <cstddef>

// Problem constants (match reference)
#define B_   32
#define S_   128
#define H_   1024
#define I_   512
#define V_   32000
#define T_   64
#define G4H  4096   // 4*H

typedef __attribute__((ext_vector_type(16))) __bf16 v16bf;
typedef __attribute__((ext_vector_type(8)))  float  v8f;
typedef unsigned short u16;

struct Q32 { uint4 a, b; };

__device__ __forceinline__ u16 f2bf(float f) {
  unsigned u = __builtin_bit_cast(unsigned, f);
  u += 0x7FFFu + ((u >> 16) & 1u);          // round-to-nearest-even
  return (u16)(u >> 16);
}

// A-matrix 16x32 bf16 fragment (ISA 7.12.2): lanes 0-15 row=l16 K=k0..k0+7,k0+16..23;
// lanes 16-31 same rows, K shifted by 8. Caller passes p = &A[row*ld + k0 + half*8].
__device__ __forceinline__ v16bf load_a_frag(const u16* p) {
  Q32 q;
  q.a = *reinterpret_cast<const uint4*>(p);
  q.b = *reinterpret_cast<const uint4*>(p + 16);
  return __builtin_bit_cast(v16bf, q);
}
// B-matrix 32x16 bf16 fragment: lane n holds column n; lanes 0-15 K=k0..k0+7,k0+16..23;
// lanes 16-31 K shifted by 8 (same as A). p = &W[n*ld + k0 + half*8].
__device__ __forceinline__ v16bf load_b_frag(const u16* p) {
  Q32 q;
  q.a = *reinterpret_cast<const uint4*>(p);
  q.b = *reinterpret_cast<const uint4*>(p + 16);
  return __builtin_bit_cast(v16bf, q);
}

__device__ __forceinline__ v8f wmma_bf16(v16bf a, v16bf b, v8f c) {
  return __builtin_amdgcn_wmma_f32_16x16x32_bf16(
      /*neg_a=*/false, a, /*neg_b=*/false, b,
      /*c_mod=*/(short)0, c, /*reuse_a=*/false, /*reuse_b=*/false);
}

__device__ __forceinline__ float sigmoidf_(float x) {
  return 1.0f / (1.0f + __expf(-x));
}

// ---------------------------------------------------------------------------
// f32 -> bf16 strided convert (weight repack). grid = rows, block = 128.
__global__ void cvt_bf16_2d(const float* __restrict__ src, u16* __restrict__ dst,
                            int cols, int dld, int dcoff) {
  int r = blockIdx.x;
  const float* s = src + (size_t)r * cols;
  u16* d = dst + (size_t)r * dld + dcoff;
  for (int c = threadIdx.x; c < cols; c += blockDim.x) d[c] = f2bf(s[c]);
}

// bf16 2D copy. grid = rows, block = 256.
__global__ void copy_bf16_2d(const u16* __restrict__ src, int sld,
                             u16* __restrict__ dst, int dld, int cols) {
  int r = blockIdx.x;
  for (int c = threadIdx.x; c < cols; c += blockDim.x)
    dst[(size_t)r * dld + c] = src[(size_t)r * sld + c];
}

// Encoder embedding gather -> time-major bf16 [S*B, I]. grid = S*B, block = 128.
__global__ void enc_embed_kernel(const int* __restrict__ x, const float* __restrict__ emb,
                                 u16* __restrict__ xs) {
  int rid = blockIdx.x;             // row = s*B + b
  int s = rid >> 5, b = rid & 31;
  int tok = x[b * S_ + s];
  const float* e = emb + (size_t)tok * I_;
  u16* d = xs + (size_t)rid * I_;
  for (int i = threadIdx.x; i < I_; i += blockDim.x) d[i] = f2bf(e[i]);
}

// Decoder embedding gather into xcat[:, 0:1024] (ld 2048). grid = B, block = 256.
__global__ void dec_embed_kernel(const float* __restrict__ emb, const int* __restrict__ tok,
                                 int use_bos, u16* __restrict__ dst) {
  int b = blockIdx.x;
  int tk = use_bos ? 1 : tok[b];    // BOS = 1
  const float* e = emb + (size_t)tk * H_;
  u16* d = dst + (size_t)b * 2048;
  for (int i = threadIdx.x; i < H_; i += blockDim.x) d[i] = f2bf(e[i]);
}

// ---------------------------------------------------------------------------
// Big NT GEMM: out[m,n] = sum_k A[m,k]*W[n,k] + bias[n].
// A bf16 [M,K] (M = gridDim.y*128), W bf16 [4096,K], out f32 [M,4096].
// block = 256 (8 waves); block tile 128x64, wave tile 32x32 (2x2 WMMA tiles).
__global__ void gemm_nt_bias_kernel(const u16* __restrict__ A, int K,
                                    const u16* __restrict__ W,
                                    const float* __restrict__ bias,
                                    float* __restrict__ out) {
  int wave = threadIdx.x >> 5, lane = threadIdx.x & 31;
  int half = lane >> 4, l16 = lane & 15;
  int wy = wave >> 1, wx = wave & 1;
  int m0 = blockIdx.y * 128 + wy * 32;
  int n0 = blockIdx.x * 64 + wx * 32;
  v8f acc[2][2] = {};
  for (int k0 = 0; k0 < K; k0 += 32) {
    v16bf a0 = load_a_frag(A + (size_t)(m0 + l16) * K + k0 + half * 8);
    v16bf a1 = load_a_frag(A + (size_t)(m0 + 16 + l16) * K + k0 + half * 8);
    v16bf b0 = load_b_frag(W + (size_t)(n0 + l16) * K + k0 + half * 8);
    v16bf b1 = load_b_frag(W + (size_t)(n0 + 16 + l16) * K + k0 + half * 8);
    acc[0][0] = wmma_bf16(a0, b0, acc[0][0]);
    acc[0][1] = wmma_bf16(a0, b1, acc[0][1]);
    acc[1][0] = wmma_bf16(a1, b0, acc[1][0]);
    acc[1][1] = wmma_bf16(a1, b1, acc[1][1]);
  }
  #pragma unroll
  for (int mt = 0; mt < 2; ++mt)
    #pragma unroll
    for (int nt = 0; nt < 2; ++nt)
      #pragma unroll
      for (int r = 0; r < 8; ++r) {
        int m = m0 + mt * 16 + r + half * 8;
        int n = n0 + nt * 16 + l16;
        out[(size_t)m * G4H + n] = acc[mt][nt][r] + bias[n];
      }
}

// ---------------------------------------------------------------------------
// Fused LSTM cell step with intra-block K-split x4.
// gates[b,n] = A[b,:]·W[n,:] (+ addf[b,n] | + bias[n]), then i,f,g,o
// nonlinearity, c update (f32, in place), h written as bf16 to up to two
// destinations. A bf16 [32,K]; W bf16 [4096,K] (torch gate order i,f,g,o in
// blocks of H along N).
// grid = 16, block = 512 (16 waves), dynamic LDS = 12*64*32*4 = 96 KB.
// Wave w: colgrp = w&3 owns 16 cols of H (across all 4 gates, both 16-row M
// tiles); kslice = w>>2 covers K/4. Partials from kslices 1..3 go through LDS
// and are summed by kslice 0 in a FIXED order (deterministic, no atomics).
#define CELL_THREADS 512
#define CELL_LDS_BYTES (12 * 64 * 32 * 4)
__global__ void lstm_cell_kernel(const u16* __restrict__ A, int lda, int K,
                                 const u16* __restrict__ W,
                                 const float* __restrict__ addf,   // [32,4096] or null
                                 const float* __restrict__ bias,   // [4096] or null
                                 float* __restrict__ c,            // [32,1024] f32 in/out
                                 u16* __restrict__ h0p, int ldh0,
                                 u16* __restrict__ h1p, int ldh1) {
  extern __shared__ float red[];    // [12 regions][64 slots][32 lanes]
  int wave = threadIdx.x >> 5, lane = threadIdx.x & 31;
  int half = lane >> 4, l16 = lane & 15;
  int colgrp = wave & 3, kslice = wave >> 2;
  int hcol = blockIdx.x * 64 + colgrp * 16;
  int Ks = K >> 2;
  int kbeg = kslice * Ks, kend = kbeg + Ks;
  v8f acc[2][4] = {};
  for (int k0 = kbeg; k0 < kend; k0 += 32) {
    v16bf a0 = load_a_frag(A + (size_t)(l16) * lda + k0 + half * 8);
    v16bf a1 = load_a_frag(A + (size_t)(16 + l16) * lda + k0 + half * 8);
    #pragma unroll
    for (int g = 0; g < 4; ++g) {
      const u16* wp = W + (size_t)(g * H_ + hcol + l16) * K + k0 + half * 8;
      if (k0 + 32 < kend) __builtin_prefetch(wp + 32, 0, 1);   // global_prefetch_b8
      v16bf bf = load_b_frag(wp);
      acc[0][g] = wmma_bf16(a0, bf, acc[0][g]);
      acc[1][g] = wmma_bf16(a1, bf, acc[1][g]);
    }
  }
  if (kslice != 0) {
    float* rp = red + ((kslice - 1) * 4 + colgrp) * 64 * 32;
    #pragma unroll
    for (int mt = 0; mt < 2; ++mt)
      #pragma unroll
      for (int g = 0; g < 4; ++g)
        #pragma unroll
        for (int r = 0; r < 8; ++r)
          rp[((mt * 4 + g) * 8 + r) * 32 + lane] = acc[mt][g][r];
  }
  __syncthreads();
  if (kslice != 0) return;
  #pragma unroll
  for (int s = 1; s <= 3; ++s) {            // fixed order -> bitwise deterministic
    const float* rp = red + ((s - 1) * 4 + colgrp) * 64 * 32;
    #pragma unroll
    for (int mt = 0; mt < 2; ++mt)
      #pragma unroll
      for (int g = 0; g < 4; ++g)
        #pragma unroll
        for (int r = 0; r < 8; ++r)
          acc[mt][g][r] += rp[((mt * 4 + g) * 8 + r) * 32 + lane];
  }
  #pragma unroll
  for (int mt = 0; mt < 2; ++mt)
    #pragma unroll
    for (int r = 0; r < 8; ++r) {
      int row = mt * 16 + r + half * 8;   // batch row 0..31
      int col = hcol + l16;               // hidden index 0..1023
      float gi = acc[mt][0][r], gf = acc[mt][1][r];
      float gg = acc[mt][2][r], go = acc[mt][3][r];
      if (addf) {
        const float* ap = addf + (size_t)row * G4H;
        gi += ap[col]; gf += ap[H_ + col]; gg += ap[2 * H_ + col]; go += ap[3 * H_ + col];
      }
      if (bias) {
        gi += bias[col]; gf += bias[H_ + col]; gg += bias[2 * H_ + col]; go += bias[3 * H_ + col];
      }
      float ig = sigmoidf_(gi), fg = sigmoidf_(gf), og = sigmoidf_(go);
      float gv = tanhf(gg);
      size_t ci = (size_t)row * H_ + col;
      float cn = fg * c[ci] + ig * gv;
      c[ci] = cn;
      u16 hb = f2bf(og * tanhf(cn));
      h0p[(size_t)row * ldh0 + col] = hb;
      if (h1p) h1p[(size_t)row * ldh1 + col] = hb;
    }
}

// ---------------------------------------------------------------------------
// Prediction logits: out[b, t, 0, n] = A[b,:]·predW[n,:] + predb[n].
// A bf16 [32, H] with row stride lda; W bf16 [V, H]. grid = V/64, block = 128.
__global__ void logits_kernel(const u16* __restrict__ A, int lda,
                              const u16* __restrict__ W,
                              const float* __restrict__ bias,
                              float* __restrict__ out, int t) {
  int wave = threadIdx.x >> 5, lane = threadIdx.x & 31;
  int half = lane >> 4, l16 = lane & 15;
  int n0 = blockIdx.x * 64 + wave * 16;
  v8f acc[2] = {};
  for (int k0 = 0; k0 < H_; k0 += 32) {
    v16bf a0 = load_a_frag(A + (size_t)(l16) * lda + k0 + half * 8);
    v16bf a1 = load_a_frag(A + (size_t)(16 + l16) * lda + k0 + half * 8);
    v16bf bf = load_b_frag(W + (size_t)(n0 + l16) * H_ + k0 + half * 8);
    acc[0] = wmma_bf16(a0, bf, acc[0]);
    acc[1] = wmma_bf16(a1, bf, acc[1]);
  }
  int n = n0 + l16;
  float bv = bias[n];
  #pragma unroll
  for (int mt = 0; mt < 2; ++mt)
    #pragma unroll
    for (int r = 0; r < 8; ++r) {
      int row = mt * 16 + r + half * 8;
      out[(size_t)row * T_ * V_ + (size_t)t * V_ + n] = acc[mt][r] + bv;
    }
}

// Greedy argmax over V per batch row (first max, like jnp.argmax).
// grid = B, block = 256.
__global__ void argmax_kernel(const float* __restrict__ out, int t, int* __restrict__ tok) {
  __shared__ float sv[256];
  __shared__ int   si[256];
  int b = blockIdx.x;
  const float* row = out + (size_t)b * T_ * V_ + (size_t)t * V_;
  float best = -3.4e38f; int bi = 0;
  for (int v = threadIdx.x; v < V_; v += 256) {
    float val = row[v];
    if (val > best) { best = val; bi = v; }   // strided ascending -> first max per thread
  }
  sv[threadIdx.x] = best; si[threadIdx.x] = bi;
  __syncthreads();
  for (int s = 128; s > 0; s >>= 1) {
    if ((int)threadIdx.x < s) {
      float ov = sv[threadIdx.x + s]; int oi = si[threadIdx.x + s];
      if (ov > sv[threadIdx.x] || (ov == sv[threadIdx.x] && oi < si[threadIdx.x])) {
        sv[threadIdx.x] = ov; si[threadIdx.x] = oi;
      }
    }
    __syncthreads();
  }
  if (threadIdx.x == 0) tok[b] = si[0];
}

// ---------------------------------------------------------------------------
extern "C" void kernel_launch(void* const* d_in, const int* in_sizes, int n_in,
                              void* d_out, int out_size, void* d_ws, size_t ws_size,
                              hipStream_t stream) {
  (void)in_sizes; (void)n_in; (void)out_size; (void)ws_size;
  const int*   x       = (const int*)  d_in[0];
  const float* enc_emb = (const float*)d_in[1];
  const float* dec_emb = (const float*)d_in[2];
  const float* eWih0   = (const float*)d_in[3];
  const float* eWhh0   = (const float*)d_in[4];
  const float* eb0     = (const float*)d_in[5];
  const float* eWih1   = (const float*)d_in[6];
  const float* eWhh1   = (const float*)d_in[7];
  const float* eb1     = (const float*)d_in[8];
  const float* dWih0   = (const float*)d_in[9];
  const float* dWhh0   = (const float*)d_in[10];
  const float* db0     = (const float*)d_in[11];
  const float* dWih1   = (const float*)d_in[12];
  const float* dWhh1   = (const float*)d_in[13];
  const float* db1     = (const float*)d_in[14];
  const float* predW   = (const float*)d_in[15];
  const float* predb   = (const float*)d_in[16];
  float* out = (float*)d_out;

  // ---- workspace layout (~200 MB) ----
  char* ws = (char*)d_ws;
  size_t off = 0;
  auto alloc = [&](size_t bytes) -> char* {
    off = (off + 255) & ~(size_t)255;
    char* p = ws + off;
    off += bytes;
    return p;
  };
  u16* xs      = (u16*)alloc((size_t)S_ * B_ * I_ * 2);   // enc embeddings, time-major
  u16* w_eih0  = (u16*)alloc((size_t)G4H * I_ * 2);
  u16* w_ehh0  = (u16*)alloc((size_t)G4H * H_ * 2);
  u16* w_eih1  = (u16*)alloc((size_t)G4H * H_ * 2);
  u16* w_ehh1  = (u16*)alloc((size_t)G4H * H_ * 2);
  u16* w_dcat0 = (u16*)alloc((size_t)G4H * 2048 * 2);     // [Wih0 | Whh0]
  u16* w_dcat1 = (u16*)alloc((size_t)G4H * 2048 * 2);     // [Wih1 | Whh1]
  u16* w_pred  = (u16*)alloc((size_t)V_ * H_ * 2);
  u16* ys0     = (u16*)alloc((size_t)S_ * B_ * H_ * 2);   // layer-0 outputs (bf16)
  u16* hb0a    = (u16*)alloc((size_t)B_ * H_ * 2);
  u16* hb0b    = (u16*)alloc((size_t)B_ * H_ * 2);
  u16* hb1a    = (u16*)alloc((size_t)B_ * H_ * 2);
  u16* hb1b    = (u16*)alloc((size_t)B_ * H_ * 2);
  u16* xc0a    = (u16*)alloc((size_t)B_ * 2048 * 2);      // decoder [e | h0]
  u16* xc0b    = (u16*)alloc((size_t)B_ * 2048 * 2);
  u16* xc1a    = (u16*)alloc((size_t)B_ * 2048 * 2);      // decoder [h0 | h1]
  u16* xc1b    = (u16*)alloc((size_t)B_ * 2048 * 2);
  float* Xg    = (float*)alloc((size_t)S_ * B_ * G4H * 4); // precomputed x@Wih+b (reused)
  float* c0    = (float*)alloc((size_t)B_ * H_ * 4);
  float* c1    = (float*)alloc((size_t)B_ * H_ * 4);
  int*   tok   = (int*)alloc((size_t)B_ * 4);

  // ---- weight repack to bf16 (once per launch; no persistent state) ----
  cvt_bf16_2d<<<G4H, 128, 0, stream>>>(eWih0, w_eih0, I_, I_, 0);
  cvt_bf16_2d<<<G4H, 128, 0, stream>>>(eWhh0, w_ehh0, H_, H_, 0);
  cvt_bf16_2d<<<G4H, 128, 0, stream>>>(eWih1, w_eih1, H_, H_, 0);
  cvt_bf16_2d<<<G4H, 128, 0, stream>>>(eWhh1, w_ehh1, H_, H_, 0);
  cvt_bf16_2d<<<G4H, 128, 0, stream>>>(dWih0, w_dcat0, H_, 2048, 0);
  cvt_bf16_2d<<<G4H, 128, 0, stream>>>(dWhh0, w_dcat0, H_, 2048, 1024);
  cvt_bf16_2d<<<G4H, 128, 0, stream>>>(dWih1, w_dcat1, H_, 2048, 0);
  cvt_bf16_2d<<<G4H, 128, 0, stream>>>(dWhh1, w_dcat1, H_, 2048, 1024);
  cvt_bf16_2d<<<V_,  128, 0, stream>>>(predW, w_pred,  H_, H_, 0);

  enc_embed_kernel<<<S_ * B_, 128, 0, stream>>>(x, enc_emb, xs);

  hipMemsetAsync(hb0a, 0, (size_t)B_ * H_ * 2, stream);
  hipMemsetAsync(hb1a, 0, (size_t)B_ * H_ * 2, stream);
  hipMemsetAsync(c0,   0, (size_t)B_ * H_ * 4, stream);
  hipMemsetAsync(c1,   0, (size_t)B_ * H_ * 4, stream);

  // ---- encoder layer 0: bulk input GEMM then 128 recurrent steps ----
  gemm_nt_bias_kernel<<<dim3(64, 32), 256, 0, stream>>>(xs, I_, w_eih0, eb0, Xg);
  u16* hb0[2] = {hb0a, hb0b};
  for (int t = 0; t < S_; ++t)
    lstm_cell_kernel<<<16, CELL_THREADS, CELL_LDS_BYTES, stream>>>(
        hb0[t & 1], H_, H_, w_ehh0, Xg + (size_t)t * B_ * G4H, nullptr, c0,
        hb0[(t + 1) & 1], H_, ys0 + (size_t)t * B_ * H_, H_);

  // ---- encoder layer 1 ----
  gemm_nt_bias_kernel<<<dim3(64, 32), 256, 0, stream>>>(ys0, H_, w_eih1, eb1, Xg);
  u16* hb1[2] = {hb1a, hb1b};
  for (int t = 0; t < S_; ++t)
    lstm_cell_kernel<<<16, CELL_THREADS, CELL_LDS_BYTES, stream>>>(
        hb1[t & 1], H_, H_, w_ehh1, Xg + (size_t)t * B_ * G4H, nullptr, c1,
        hb1[(t + 1) & 1], H_, nullptr, 0);

  // ---- seed decoder recurrent inputs with encoder final hidden states ----
  copy_bf16_2d<<<B_, 256, 0, stream>>>(ys0 + (size_t)(S_ - 1) * B_ * H_, H_,
                                       xc0a + 1024, 2048, H_);             // h0T
  copy_bf16_2d<<<B_, 256, 0, stream>>>(hb1[0], H_, xc1a + 1024, 2048, H_); // h1T
  // c0/c1 already hold cT from the encoder loops.

  // ---- decoder: 64 greedy steps ----
  u16* xc0[2] = {xc0a, xc0b};
  u16* xc1[2] = {xc1a, xc1b};
  for (int t = 0; t < T_; ++t) {
    dec_embed_kernel<<<B_, 256, 0, stream>>>(dec_emb, tok, t == 0 ? 1 : 0, xc0[t & 1]);
    // cell 0: reads [e|h0_prev]; writes h0_new -> xc1_cur[:,0:H] and xc0_next[:,H:2H]
    lstm_cell_kernel<<<16, CELL_THREADS, CELL_LDS_BYTES, stream>>>(
        xc0[t & 1], 2048, 2048, w_dcat0, nullptr, db0, c0,
        xc1[t & 1], 2048, xc0[(t + 1) & 1] + 1024, 2048);
    // cell 1: reads [h0_new|h1_prev]; writes h1_new -> xc1_next[:,H:2H]
    lstm_cell_kernel<<<16, CELL_THREADS, CELL_LDS_BYTES, stream>>>(
        xc1[t & 1], 2048, 2048, w_dcat1, nullptr, db1, c1,
        xc1[(t + 1) & 1] + 1024, 2048, nullptr, 0);
    logits_kernel<<<V_ / 64, 128, 0, stream>>>(
        xc1[(t + 1) & 1] + 1024, 2048, w_pred, predb, out, t);
    argmax_kernel<<<B_, 256, 0, stream>>>(out, t, tok);
  }
}
